// MultiLoraLinear_87978110091947
// MI455X (gfx1250) — matrix operand order repaired and checked by reading, not compiled, not run on hardware
//
#include <hip/hip_runtime.h>

// ---------------- problem constants (from reference) ----------------
#define BATCH   8
#define SEQ     2048
#define DIN     4096
#define DOUT    4096
#define NLORA   3
#define RANK    16
#define MTOT    (BATCH * SEQ)      // 16384
#define LORA_SCALE 1.0f            // alpha/rank = 16/16

// ---------------- tiling ----------------
#define TILE_M  128
#define TILE_N  128
#define TILE_K  32
#define LDSK    40                           // padded LDS row stride in bf16 elems (80B: 16B-aligned, conflict-free)
#define PLANE   (TILE_M * LDSK)              // ushorts per plane
#define SMEM_MAIN_BYTES (4 * PLANE * 2)      // Ahi,Alo,Bhi,Blo -> 40960 bytes

typedef __bf16 bf16_t;
typedef bf16_t v16bf __attribute__((ext_vector_type(16)));
typedef bf16_t v8bf  __attribute__((ext_vector_type(8)));
typedef float  v8f   __attribute__((ext_vector_type(8)));

__device__ __forceinline__ v16bf cat16(v8bf a, v8bf b) {
  return __builtin_shufflevector(a, b, 0,1,2,3,4,5,6,7,8,9,10,11,12,13,14,15);
}

// A-matrix 16x32 bf16 fragment (ISA 7.12.2): lanes 0-15: V0-3=K0..7, V4-7=K16..23
//                                            lanes 16-31: V0-3=K8..15, V4-7=K24..31
__device__ __forceinline__ v16bf frag_a(const unsigned short* plane, int row0, int lane) {
  const int m  = lane & 15;
  const int hi = lane >> 4;
  const unsigned short* p = plane + (row0 + m) * LDSK;
  v8bf f0 = *(const v8bf*)(p + (hi ? 8  : 0));
  v8bf f1 = *(const v8bf*)(p + (hi ? 24 : 16));
  return cat16(f0, f1);
}

// B-matrix 32x16 bf16 fragment: lanes 0-15 hold K=0..7,16..23 of column n=lane,
// lanes 16-31 hold K=8..15,24..31 of column n=lane-16 (columns of B = rows of W)
__device__ __forceinline__ v16bf frag_b(const unsigned short* plane, int row0, int lane) {
  const int n  = lane & 15;
  const int hi = lane >> 4;
  const unsigned short* p = plane + (row0 + n) * LDSK;
  v8bf f0 = *(const v8bf*)(p + (hi ? 8  : 0));
  v8bf f1 = *(const v8bf*)(p + (hi ? 24 : 16));
  return cat16(f0, f1);
}

// fp32 -> (hi bf16 = truncation, lo bf16 = RNE(residual)); hi+lo carries ~16 mantissa bits
__device__ __forceinline__ void split_f32(float f, unsigned short& h, unsigned short& l) {
  unsigned u = __builtin_bit_cast(unsigned, f);
  h = (unsigned short)(u >> 16);
  float hf = __builtin_bit_cast(float, u & 0xFFFF0000u);
  float lof = f - hf;
  l = __builtin_bit_cast(unsigned short, (bf16_t)lof);
}

__device__ __forceinline__ void store8u(unsigned short* dst, const unsigned short* s) {
  uint4 v;
  v.x = (unsigned)s[0] | ((unsigned)s[1] << 16);
  v.y = (unsigned)s[2] | ((unsigned)s[3] << 16);
  v.z = (unsigned)s[4] | ((unsigned)s[5] << 16);
  v.w = (unsigned)s[6] | ((unsigned)s[7] << 16);
  *(uint4*)dst = v;
}

// segment -> adapter resolution (searchsorted(starts, b, 'right') - 1)
__device__ __forceinline__ void resolve_adapter(const long long* starts, int b,
                                                bool& use_lora, int& aidx) {
  int cnt = 0;
#pragma unroll
  for (int i = 0; i < 4; ++i) cnt += (starts[i] <= (long long)b) ? 1 : 0;
  const int seg = cnt - 1;
  use_lora = seg > 0;
  int a = seg - 1;
  if (a < 0) a = 0;
  if (a > NLORA - 1) a = NLORA - 1;
  aidx = a;
}

// ------------------------------------------------------------------
// Pre-kernel: xa[m, r] = sum_d x[m,d] * A[adapter(m), r, d]   (zero if no LoRA)
// One block of 256 threads per 16 rows; thread (row = t>>4, r = t&15).
// ------------------------------------------------------------------
__global__ __launch_bounds__(256)
void mlora_xa(const float* __restrict__ x, const float* __restrict__ loraA,
              const long long* __restrict__ starts, float* __restrict__ xa)
{
  __shared__ float xs[16][132];   // +4 pad: A-plane reads stride 132 -> conflict-free
  __shared__ float as[16][132];
  const int tid = threadIdx.x;
  const int row = tid >> 4;       // 0..15
  const int rr  = tid & 15;       // 0..15
  const int m0  = blockIdx.x * 16;

  bool use_lora; int aidx;
  resolve_adapter(starts, m0 / SEQ, use_lora, aidx);

  const float* Ab = loraA + (size_t)aidx * RANK * DIN;
  float acc = 0.f;
  for (int k0 = 0; k0 < DIN; k0 += 128) {
    const int lc = rr * 8;
    const float4* pxs = (const float4*)(x + (size_t)(m0 + row) * DIN + k0 + lc);
    const float4* pas = (const float4*)(Ab + (size_t)row * DIN + k0 + lc);
    *(float4*)&xs[row][lc]     = pxs[0];
    *(float4*)&xs[row][lc + 4] = pxs[1];
    *(float4*)&as[row][lc]     = pas[0];
    *(float4*)&as[row][lc + 4] = pas[1];
    __syncthreads();
#pragma unroll 16
    for (int c = 0; c < 128; ++c) acc += xs[row][c] * as[rr][c];
    __syncthreads();
  }
  xa[(size_t)(m0 + row) * RANK + rr] = use_lora ? acc * LORA_SCALE : 0.f;
}

// ------------------------------------------------------------------
// Main kernel: 128x128 C tile, bf16x3 split-precision WMMA GEMM,
// fused bias + rank-16 LoRA epilogue.
// ------------------------------------------------------------------
__global__ __launch_bounds__(256)
void mlora_gemm(const float* __restrict__ x, const float* __restrict__ w,
                const float* __restrict__ bias, const float* __restrict__ loraB,
                const long long* __restrict__ starts, const float* __restrict__ xa,
                float* __restrict__ out)
{
  extern __shared__ __align__(16) unsigned char smem_raw[];
  unsigned short* sAhi = (unsigned short*)smem_raw;
  unsigned short* sAlo = sAhi + PLANE;
  unsigned short* sBhi = sAlo + PLANE;
  unsigned short* sBlo = sBhi + PLANE;

  const int tid  = threadIdx.x;
  const int lane = tid & 31;
  const int wave = tid >> 5;
  const int wm   = wave >> 2;   // 0..1 : 64-row slab
  const int wn   = wave & 3;    // 0..3 : 32-col slab

  const int m0 = blockIdx.y * TILE_M;   // 2048 % 128 == 0 -> tile within one sample
  const int n0 = blockIdx.x * TILE_N;

  bool use_lora; int aidx;
  resolve_adapter(starts, m0 / SEQ, use_lora, aidx);

  // global staging: thread t loads 16 consecutive floats of one row for x and for W
  const int lrow = tid >> 1;            // 0..127
  const int lcol = (tid & 1) * 16;      // 0 / 16
  const float* px = x + (size_t)(m0 + lrow) * DIN + lcol;
  const float* pw = w + (size_t)(n0 + lrow) * DIN + lcol;

  float4 rx[4], rw[4];
#pragma unroll
  for (int q = 0; q < 4; ++q) { rx[q] = ((const float4*)px)[q]; rw[q] = ((const float4*)pw)[q]; }

  v8f acc[4][2];
#pragma unroll
  for (int i = 0; i < 4; ++i) {
#pragma unroll
    for (int j = 0; j < 2; ++j) { v8f z = {}; acc[i][j] = z; }
  }

  const int NK = DIN / TILE_K;   // 128
  for (int kk = 0; kk < NK; ++kk) {
    // split current fp32 slab into hi/lo bf16 planes and store to LDS
    {
      unsigned short hx[16], lx[16], hw[16], lw[16];
#pragma unroll
      for (int q = 0; q < 4; ++q) {
        float fx[4] = {rx[q].x, rx[q].y, rx[q].z, rx[q].w};
        float fw[4] = {rw[q].x, rw[q].y, rw[q].z, rw[q].w};
#pragma unroll
        for (int e = 0; e < 4; ++e) {
          split_f32(fx[e], hx[q * 4 + e], lx[q * 4 + e]);
          split_f32(fw[e], hw[q * 4 + e], lw[q * 4 + e]);
        }
      }
      unsigned short* pa;
      pa = sAhi + lrow * LDSK + lcol; store8u(pa, hx); store8u(pa + 8, hx + 8);
      pa = sAlo + lrow * LDSK + lcol; store8u(pa, lx); store8u(pa + 8, lx + 8);
      pa = sBhi + lrow * LDSK + lcol; store8u(pa, hw); store8u(pa + 8, hw + 8);
      pa = sBlo + lrow * LDSK + lcol; store8u(pa, lw); store8u(pa + 8, lw + 8);
    }
    __syncthreads();

    // prefetch next K slab into registers (hides global latency behind WMMAs)
    if (kk + 1 < NK) {
      const float* nx = px + (size_t)(kk + 1) * TILE_K;
      const float* nw = pw + (size_t)(kk + 1) * TILE_K;
#pragma unroll
      for (int q = 0; q < 4; ++q) { rx[q] = ((const float4*)nx)[q]; rw[q] = ((const float4*)nw)[q]; }
    }

    // 24 WMMAs per wave per K step: hi*hi + lo*hi + hi*lo  (bf16x3)
    v16bf bh0 = frag_b(sBhi, wn * 32 + 0,  lane);
    v16bf bh1 = frag_b(sBhi, wn * 32 + 16, lane);
    v16bf bl0 = frag_b(sBlo, wn * 32 + 0,  lane);
    v16bf bl1 = frag_b(sBlo, wn * 32 + 16, lane);
#pragma unroll
    for (int i = 0; i < 4; ++i) {
      v16bf ah = frag_a(sAhi, wm * 64 + i * 16, lane);
      v16bf al = frag_a(sAlo, wm * 64 + i * 16, lane);
      acc[i][0] = __builtin_amdgcn_wmma_f32_16x16x32_bf16(false, ah, false, bh0, (short)0, acc[i][0], false, false);
      acc[i][1] = __builtin_amdgcn_wmma_f32_16x16x32_bf16(false, ah, false, bh1, (short)0, acc[i][1], false, false);
      acc[i][0] = __builtin_amdgcn_wmma_f32_16x16x32_bf16(false, al, false, bh0, (short)0, acc[i][0], false, false);
      acc[i][1] = __builtin_amdgcn_wmma_f32_16x16x32_bf16(false, al, false, bh1, (short)0, acc[i][1], false, false);
      acc[i][0] = __builtin_amdgcn_wmma_f32_16x16x32_bf16(false, ah, false, bl0, (short)0, acc[i][0], false, false);
      acc[i][1] = __builtin_amdgcn_wmma_f32_16x16x32_bf16(false, ah, false, bl1, (short)0, acc[i][1], false, false);
    }
    __syncthreads();
  }

  // ---- epilogue: bias + rank-16 LoRA correction (reuse LDS) ----
  float* sbw = (float*)smem_raw;                                   // [128][20] (pad: conflict-free)
  float* sxa = (float*)(smem_raw + TILE_N * 20 * sizeof(float));   // [128][16]
  if (use_lora) {
    const int nr = tid >> 1;
    const int rc = (tid & 1) * 8;
    const float* pb  = loraB + ((size_t)aidx * DOUT + n0 + nr) * RANK + rc;
    const float* pxa = xa + (size_t)(m0 + nr) * RANK + rc;
#pragma unroll
    for (int e = 0; e < 8; ++e) {
      sbw[nr * 20 + rc + e] = pb[e];
      sxa[nr * 16 + rc + e] = pxa[e];
    }
  }
  __syncthreads();

  // C/D layout: VGPR v holds M = v (+8 for lanes 16-31); N = lane & 15
  const int lm_hi = (lane >> 4) * 8;
  const int ln    = lane & 15;
#pragma unroll
  for (int j = 0; j < 2; ++j) {
    const int ncol = wn * 32 + j * 16 + ln;
    const float bv = bias[n0 + ncol];
#pragma unroll
    for (int i = 0; i < 4; ++i) {
      const int mbase = wm * 64 + i * 16 + lm_hi;
#pragma unroll
      for (int v = 0; v < 8; ++v) {
        const int m = mbase + v;
        float s = bv;
        if (use_lora) {
#pragma unroll
          for (int r = 0; r < RANK; ++r) s += sxa[m * 16 + r] * sbw[ncol * 20 + r];
        }
        out[(size_t)(m0 + m) * DOUT + (n0 + ncol)] = acc[i][j][v] + s;
      }
    }
  }
}

// ------------------------------------------------------------------
extern "C" void kernel_launch(void* const* d_in, const int* in_sizes, int n_in,
                              void* d_out, int out_size, void* d_ws, size_t ws_size,
                              hipStream_t stream) {
  (void)in_sizes; (void)n_in; (void)out_size; (void)ws_size;
  const float*     x      = (const float*)d_in[0];
  const float*     weight = (const float*)d_in[1];
  const float*     bias   = (const float*)d_in[2];
  const float*     loraA  = (const float*)d_in[3];
  const float*     loraB  = (const float*)d_in[4];
  const long long* starts = (const long long*)d_in[5];  // int64
  float* out = (float*)d_out;
  float* xa  = (float*)d_ws;   // MTOT * RANK floats = 1 MB scratch

  mlora_xa<<<MTOT / 16, 256, 0, stream>>>(x, loraA, starts, xa);
  mlora_gemm<<<dim3(DOUT / TILE_N, MTOT / TILE_M), 256, SMEM_MAIN_BYTES, stream>>>(
      x, weight, bias, loraB, starts, xa, out);
}
